// RNN_10977936409116
// MI455X (gfx1250) — matrix operand order repaired and checked by reading, not compiled, not run on hardware
//
#include <hip/hip_runtime.h>
#include <math.h>

typedef __attribute__((ext_vector_type(16))) _Float16 v16h;
typedef __attribute__((ext_vector_type(8)))  float    v8f;

#define B_SZ 1024
#define T_SZ 1024
#define N_SZ 256

constexpr float DT      = 0.2f;          // DT_OVER_TAU
constexpr float DECAY   = 1.0f - DT;     // 0.8
constexpr float INV_N   = 1.0f / 256.0f;
constexpr float SCALE_W = DT * INV_N;    // folded into W' = DT/N * (n @ m^T)

// LDS layout. W: B-fragments of W' (f16), [kt(8)][nt(16)] blocks of 1KB;
// within a block, lane l's 8-byte chunk c lives at c*256 + l*8 (conflict-free b64).
// R: A-fragments of r=tanh(x) (f16), double buffered, [kt(8)] blocks of 1KB, same chunking.
struct Smem {
  unsigned int W[32768];       // 128 KB
  unsigned int R[2][2048];     // 2 x 8 KB
  float su[16][68];            // u staging: 16 rows x 64 steps (+pad)
  float wiDT[256];             // DT * wi
  float wON[256];              // w / N
  float zpart[2][4][16];       // per-wave z partials, double buffered
  union {                      // prologue staging overlaid with z line buffer
    struct { float mS[512]; float nS[512]; } p;
    float zbuf[16][68];        // z staging: 16 rows x 64 steps (+pad)
  } ov;
};

__device__ __forceinline__ float fast_tanh(float x) {
#if __has_builtin(__builtin_amdgcn_tanhf)
  return __builtin_amdgcn_tanhf(x);            // v_tanh_f32 (TRANS, co-executes with WMMA)
#elif __has_builtin(__builtin_amdgcn_tanh_f32)
  return __builtin_amdgcn_tanh_f32(x);
#else
  float ax = fabsf(x);
  float e  = __expf(-2.0f * ax);               // underflows -> 0 for big ax
  float r  = (1.0f - e) * __builtin_amdgcn_rcpf(1.0f + e);
  return copysignf(r, x);
#endif
}

__global__ __launch_bounds__(128)
void rnn_lowrank_wmma(const float* __restrict__ u,  const float* __restrict__ m,
                      const float* __restrict__ n,  const float* __restrict__ wi,
                      const float* __restrict__ w,  float* __restrict__ out)
{
  extern __shared__ char smem_raw[];
  Smem& s = *reinterpret_cast<Smem*>(smem_raw);

  const int tid  = threadIdx.x;
  const int wv   = tid >> 5;        // wave 0..3
  const int lane = tid & 31;
  const int hi   = lane >> 4;       // half-wave
  const int cl   = lane & 15;       // column-in-tile / M-in-tile
  const int hi8  = hi * 8;
  const int rowbase = blockIdx.x * 16;

  // -------- prologue: constants + first u chunk + zero r_0 --------
  for (int j = tid; j < 512; j += 128) { s.ov.p.mS[j] = m[j]; s.ov.p.nS[j] = n[j]; }
  for (int j = tid; j < 256; j += 128) { s.wiDT[j] = DT * wi[j]; s.wON[j] = w[j] * INV_N; }
  for (int j = tid; j < 2048; j += 128) s.R[0][j] = 0u;   // r_0 = tanh(0) = 0
  {
    int row = tid >> 3, j0 = (tid & 7) * 8;
    const float* up = u + (size_t)(rowbase + row) * T_SZ + j0;
    #pragma unroll
    for (int q = 0; q < 8; ++q) s.su[row][j0 + q] = up[q];
  }
  __syncthreads();

  // -------- generate W' = DT/N * n @ m^T in B-fragment order (f16 pairs) --------
  // B layout (32x16, f16): lane l, element e  ->  K = kt*32 + (e>>3)*16 + (l>>4)*8 + (e&7), N = nt*16 + (l&15)
  for (int q = tid; q < 32768; q += 128) {
    int hp = q & 1, l = (q >> 1) & 31, c = (q >> 6) & 3, nt = (q >> 8) & 15, kt = q >> 12;
    int e0 = c * 4 + hp * 2;
    int K  = kt * 32 + (e0 >> 3) * 16 + (l >> 4) * 8 + (e0 & 7);
    int Nc = nt * 16 + (l & 15);
    float m0 = s.ov.p.mS[2 * Nc], m1 = s.ov.p.mS[2 * Nc + 1];
    float v0 = SCALE_W * (s.ov.p.nS[2 * K]     * m0 + s.ov.p.nS[2 * K + 1] * m1);
    float v1 = SCALE_W * (s.ov.p.nS[2 * K + 2] * m0 + s.ov.p.nS[2 * K + 3] * m1);
    union { unsigned int u32; _Float16 h[2]; } pk;
    pk.h[0] = (_Float16)v0; pk.h[1] = (_Float16)v1;
    s.W[(kt * 16 + nt) * 256 + c * 64 + l * 2 + hp] = pk.u32;
  }
  __syncthreads();

  // -------- state: x in C-fragment layout, 4 tiles of 16x16 per wave --------
  v8f x[4];
  #pragma unroll
  for (int ti = 0; ti < 4; ++ti)
    #pragma unroll
    for (int i = 0; i < 8; ++i) x[ti][i] = 0.0f;

  for (int t = 0; t < T_SZ; ++t) {
    const int cur = t & 1, nxt = cur ^ 1, tloc = t & 63;

    // A fragments of r_t (f16 16x32 layout), 4 conflict-free b64 loads each
    v16h a[8];
    {
      const uint2* Rb = (const uint2*)s.R[cur];
      #pragma unroll
      for (int kt = 0; kt < 8; ++kt) {
        union { v16h v; uint2 d[4]; } af;
        int bi = kt * 128 + lane;
        af.d[0] = Rb[bi]; af.d[1] = Rb[bi + 32]; af.d[2] = Rb[bi + 64]; af.d[3] = Rb[bi + 96];
        a[kt] = af.v;
      }
    }

    float su8[8];
    #pragma unroll
    for (int i = 0; i < 8; ++i) su8[i] = s.su[hi8 + i][tloc];

    // x_{t+1} = 0.8*x + u_t*(DT*wi) + r_t @ W'   (C init carries the elementwise part)
    const uint2* Wb = (const uint2*)s.W;
    #pragma unroll
    for (int ti = 0; ti < 4; ++ti) {
      const int nt  = wv * 4 + ti;
      const float wic = s.wiDT[nt * 16 + cl];
      v8f c;
      #pragma unroll
      for (int i = 0; i < 8; ++i) c[i] = DECAY * x[ti][i] + su8[i] * wic;
      #pragma unroll
      for (int kt = 0; kt < 8; ++kt) {
        union { v16h v; uint2 d[4]; } bf;
        int bi = (kt * 16 + nt) * 128 + lane;
        bf.d[0] = Wb[bi]; bf.d[1] = Wb[bi + 32]; bf.d[2] = Wb[bi + 64]; bf.d[3] = Wb[bi + 96];
        c = __builtin_amdgcn_wmma_f32_16x16x32_f16(false, a[kt], false, bf.v,
                                                   (short)0, c, false, false);
      }
      x[ti] = c;
    }

    // r_{t+1} = tanh(x_{t+1}) -> A-fragment LDS (next buffer) + readout partials
    float zacc[8];
    #pragma unroll
    for (int i = 0; i < 8; ++i) zacc[i] = 0.0f;
    char* Rn = (char*)s.R[nxt];
    #pragma unroll
    for (int ti = 0; ti < 4; ++ti) {
      const int col = (wv * 4 + ti) * 16 + cl;      // K index of this lane's column
      const float wc = s.wON[col];
      const int kt = col >> 5, kl = col & 31;
      const int lha = (kl >> 3) & 1, g = (kl >> 4) & 1;
      const int v  = g * 4 + ((kl >> 1) & 3);
      const int e  = 2 * v + (kl & 1);
      const int off0 = kt * 1024 + (e >> 2) * 256 + lha * 128 + (e & 3) * 2;
      #pragma unroll
      for (int i = 0; i < 8; ++i) {
        float r = fast_tanh(x[ti][i]);
        zacc[i] += r * wc;
        *(_Float16*)(Rn + off0 + (hi8 + i) * 8) = (_Float16)r;   // A layout: l = lha*16 + M
      }
    }
    #pragma unroll
    for (int i = 0; i < 8; ++i) {
      zacc[i] += __shfl_xor(zacc[i], 1, 32);
      zacc[i] += __shfl_xor(zacc[i], 2, 32);
      zacc[i] += __shfl_xor(zacc[i], 4, 32);
      zacc[i] += __shfl_xor(zacc[i], 8, 32);
    }
    if (cl == 0) {
      #pragma unroll
      for (int i = 0; i < 8; ++i) s.zpart[t & 1][wv][hi8 + i] = zacc[i];
    }

    // stage next 64 timesteps of u (overlaps with tail of compute)
    if (((t + 1) & 63) == 0 && (t + 1) < T_SZ) {
      int row = tid >> 3, j0 = (tid & 7) * 8;
      const float* up = u + (size_t)(rowbase + row) * T_SZ + (t + 1) + j0;
      #pragma unroll
      for (int q = 0; q < 8; ++q) s.su[row][j0 + q] = up[q];
    }

    __syncthreads();   // orders: R[nxt] writes, zpart writes, su staging

    if (wv == 0 && lane < 16) {
      float zs = s.zpart[t & 1][0][lane] + s.zpart[t & 1][1][lane]
               + s.zpart[t & 1][2][lane] + s.zpart[t & 1][3][lane];
      s.ov.zbuf[lane][tloc] = zs;
    }

    // flush 64 accumulated z columns with coalesced stores
    if (tloc == 63) {
      __syncthreads();   // zbuf column 63 visible to all
      int row = tid >> 3, j0 = (tid & 7) * 8;
      float* op = out + (size_t)(rowbase + row) * T_SZ + (t - 63) + j0;
      #pragma unroll
      for (int q = 0; q < 8; ++q) op[q] = s.ov.zbuf[row][j0 + q];
    }
  }
}

extern "C" void kernel_launch(void* const* d_in, const int* in_sizes, int n_in,
                              void* d_out, int out_size, void* d_ws, size_t ws_size,
                              hipStream_t stream) {
  (void)in_sizes; (void)n_in; (void)out_size; (void)d_ws; (void)ws_size;
  const float* u  = (const float*)d_in[0];
  const float* m  = (const float*)d_in[1];
  const float* n  = (const float*)d_in[2];
  const float* wi = (const float*)d_in[3];
  const float* w  = (const float*)d_in[4];
  float* out = (float*)d_out;

  size_t shmem = sizeof(Smem);
  hipFuncSetAttribute(reinterpret_cast<const void*>(rnn_lowrank_wmma),
                      hipFuncAttributeMaxDynamicSharedMemorySize, (int)shmem);
  rnn_lowrank_wmma<<<B_SZ / 16, 128, shmem, stream>>>(u, m, n, wi, w, out);
}
